// HyenaOperator_75033078661221
// MI455X (gfx1250) — matrix operand order repaired and checked by reading, not compiled, not run on hardware
//
#include <hip/hip_runtime.h>
#include <hip/hip_bf16.h>

typedef __attribute__((ext_vector_type(16))) __bf16 v16bf;
typedef __attribute__((ext_vector_type(4)))  __bf16 v4bf;
typedef __attribute__((ext_vector_type(8)))  float  v8f;
typedef __attribute__((ext_vector_type(4)))  unsigned int u32x4;
typedef __attribute__((ext_vector_type(8)))  int i32x8;
typedef __attribute__((ext_vector_type(4)))  int i32x4;

#define SEQ 8192
#define DIMV 512
#define HIDV 64

// ---------------------------------------------------------------------------
// helpers
// ---------------------------------------------------------------------------
__device__ __forceinline__ float geluf(float x) {
    return 0.5f * x * (1.0f + erff(x * 0.70710678118654752f));
}
__device__ __forceinline__ __bf16 to_bf16(float f) { return (__bf16)f; }

// ---------------------------------------------------------------------------
// TDM: load a 128-row x 32-col bf16 tile (row stride K elements) into LDS.
// D# per CDNA5 ISA:
//   group0: count=1 | lds_addr | global_addr | type=2
//   group1: data_size=1(2B), pad_enable, pad_interval=3 (16 DW), pad_amount=1
//           (2 DW)  -> LDS pitch = 32+4 = 36 bf16 (72 B, 8B-aligned rows)
//   tile_dim0=32 (contiguous K), tile_dim1=128 (rows), stride0=K
// ---------------------------------------------------------------------------
__device__ __forceinline__ void tdm_load_tile_128x32(const __bf16* gptr,
                                                     unsigned int lds_off,
                                                     int nrows, int K) {
    const unsigned long long ga = (unsigned long long)(uintptr_t)gptr;
    u32x4 g0;
    g0[0] = 1u;                                            // count=1 (valid)
    g0[1] = lds_off;                                       // lds_addr
    g0[2] = (unsigned int)ga;                              // global_addr[31:0]
    g0[3] = (unsigned int)((ga >> 32) & 0x1FFFFFFu) | (2u << 30);  // ga[56:32]|type=2
    i32x8 g1;
    g1[0] = (int)((1u << 16) | (1u << 20) | (3u << 22) | (1u << 25));
    //        data_size=2B    pad_enable   interval=16DW   amount=2DW
    g1[1] = (int)(((unsigned int)K & 0xFFFFu) << 16);                  // dim0 lo16
    g1[2] = (int)((((unsigned int)K >> 16) & 0xFFFFu) |
                  (((unsigned int)nrows & 0xFFFFu) << 16));            // dim0 hi | dim1 lo
    g1[3] = (int)((((unsigned int)nrows >> 16) & 0xFFFFu) | (32u << 16)); // dim1 hi | tile0=32
    g1[4] = (int)128u;                                                  // tile1=128, tile2=0
    g1[5] = (int)K;                                                     // stride0 lo32
    g1[6] = 0;                                                          // stride0 hi | stride1 lo
    g1[7] = 0;                                                          // stride1 hi
    i32x4 z4 = {0, 0, 0, 0};
    i32x8 z8 = {0, 0, 0, 0, 0, 0, 0, 0};
    __builtin_amdgcn_tensor_load_to_lds(g0, g1, z4, z4, z8, 0);
}

// ---------------------------------------------------------------------------
// precision conversion kernels (feed TDM with bf16 source data)
// ---------------------------------------------------------------------------
__global__ __launch_bounds__(256)
void cvt_bf16_kernel(const float* __restrict__ in, __bf16* __restrict__ out, int n4) {
    const int i = blockIdx.x * 256 + threadIdx.x;
    if (i < n4) {
        const float4 v = ((const float4*)in)[i];
        v4bf o;
        o[0] = to_bf16(v.x); o[1] = to_bf16(v.y);
        o[2] = to_bf16(v.z); o[3] = to_bf16(v.w);
        ((v4bf*)out)[i] = o;
    }
}

// in: R x C row-major f32  ->  out: C x R row-major bf16 (transposed)
__global__ void transpose_bf16_kernel(const float* __restrict__ in,
                                      __bf16* __restrict__ out, int R, int C) {
    __shared__ float tile[32][33];
    const int r0 = blockIdx.y * 32, c0 = blockIdx.x * 32;
    const int tx = threadIdx.x, ty = threadIdx.y;
    for (int i = ty; i < 32; i += 8)
        tile[i][tx] = in[(size_t)(r0 + i) * C + c0 + tx];
    __syncthreads();
    for (int i = ty; i < 32; i += 8)
        out[(size_t)(c0 + i) * R + r0 + tx] = to_bf16(tile[tx][i]);
}

// ---------------------------------------------------------------------------
// 1) Implicit filter MLP: pos(1) -> 64 -> 64 -> 512, exact GELU.
// ---------------------------------------------------------------------------
__global__ __launch_bounds__(256)
void filter_mlp_kernel(const float* __restrict__ w1, const float* __restrict__ b1,
                       const float* __restrict__ w2, const float* __restrict__ b2,
                       const float* __restrict__ w3, const float* __restrict__ b3,
                       float* __restrict__ filt_t) {
    __shared__ float h1[32][64];
    __shared__ float h2[32][64];
    const int tid = threadIdx.x;
    const int s0  = blockIdx.x * 32;
    const int p   = tid >> 3;
    const int sub = tid & 7;
    const float pos = (float)(s0 + p) / (float)(SEQ - 1);

    #pragma unroll
    for (int i = sub * 8; i < sub * 8 + 8; i++)
        h1[p][i] = geluf(pos * w1[i] + b1[i]);
    __syncthreads();

    for (int i = sub * 8; i < sub * 8 + 8; i++) {
        float acc = b2[i];
        #pragma unroll 8
        for (int j = 0; j < 64; j++) acc += h1[p][j] * w2[j * 64 + i];
        h2[p][i] = geluf(acc);
    }
    __syncthreads();

    const int sl = tid & 31;
    const int dg = tid >> 5;
    for (int k = 0; k < 64; k++) {
        const int d = dg + 8 * k;
        float acc = b3[d];
        #pragma unroll 8
        for (int j = 0; j < 64; j++) acc += h2[sl][j] * w3[j * DIMV + d];
        filt_t[(size_t)d * SEQ + s0 + sl] = acc;
    }
}

// ---------------------------------------------------------------------------
// 2) bf16 WMMA GEMM with TDM-staged, double-buffered LDS tiles.
//    C(MxN,f32) = A(MxK,bf16) * Bt(NxK,bf16)^T + bias(N)
//    Block tile 128x128, BK=32, 8 waves, wave tile 64x32 (8 accumulators).
// ---------------------------------------------------------------------------
__global__ __launch_bounds__(256)
void gemm_bf16_wmma_tdm(const __bf16* __restrict__ A, const __bf16* __restrict__ Bt,
                        const float* __restrict__ bias, float* __restrict__ C,
                        int M, int N, int K) {
    __shared__ __bf16 As[2][128][36];   // pitch 36 = TDM pad (16DW + 2DW)
    __shared__ __bf16 Bs[2][128][36];

    const int tid  = threadIdx.x;
    const int wave = tid >> 5;
    const int lane = tid & 31;
    const int m0 = blockIdx.y * 128;
    const int n0 = blockIdx.x * 128;
    const int wr = wave >> 2;
    const int wc = wave & 3;

    const unsigned int ldsA0 = (unsigned int)(uintptr_t)&As[0][0][0];
    const unsigned int ldsA1 = (unsigned int)(uintptr_t)&As[1][0][0];
    const unsigned int ldsB0 = (unsigned int)(uintptr_t)&Bs[0][0][0];
    const unsigned int ldsB1 = (unsigned int)(uintptr_t)&Bs[1][0][0];

    v8f acc[4][2];
    #pragma unroll
    for (int t = 0; t < 4; t++)
        #pragma unroll
        for (int u = 0; u < 2; u++)
            acc[t][u] = (v8f){0.f,0.f,0.f,0.f,0.f,0.f,0.f,0.f};

    // ISA fragment lane decode (CDNA5 ISA §7.12.2)
    const int aM  = lane & 15;
    const int aKb = (lane >> 4) * 8;    // A: split K run 0/8 (+16)
    const int bN  = lane & 15;
    const int bKh = (lane >> 4) * 8;    // B: split K run 0/8 (+16)

    const int nk = K / 32;

    // prologue: stage k-block 0 into buffer 0
    if (tid < 32) {
        tdm_load_tile_128x32(A  + (size_t)m0 * K, ldsA0, M, K);
        tdm_load_tile_128x32(Bt + (size_t)n0 * K, ldsB0, N, K);
    }
    __builtin_amdgcn_s_wait_tensorcnt(0);
    __syncthreads();

    for (int i = 0; i < nk; i++) {
        const int cur = i & 1;
        if (i + 1 < nk && tid < 32) {   // prefetch next k-block into other buffer
            const int k1 = (i + 1) * 32;
            tdm_load_tile_128x32(A  + (size_t)m0 * K + k1, cur ? ldsA0 : ldsA1, M, K);
            tdm_load_tile_128x32(Bt + (size_t)n0 * K + k1, cur ? ldsB0 : ldsB1, N, K);
        }

        v16bf af[4], bfr[2];
        #pragma unroll
        for (int t = 0; t < 4; t++) {
            const int row = wr * 64 + t * 16 + aM;
            #pragma unroll
            for (int i2 = 0; i2 < 4; i2++) {
                af[t][2 * i2]     = As[cur][row][aKb + 2 * i2];
                af[t][2 * i2 + 1] = As[cur][row][aKb + 2 * i2 + 1];
                af[t][2 * i2 + 8] = As[cur][row][16 + aKb + 2 * i2];
                af[t][2 * i2 + 9] = As[cur][row][16 + aKb + 2 * i2 + 1];
            }
        }
        #pragma unroll
        for (int u = 0; u < 2; u++) {
            const int col = wc * 32 + u * 16 + bN;
            #pragma unroll
            for (int i2 = 0; i2 < 16; i2++)
                bfr[u][i2] = Bs[cur][col][bKh + (i2 & 7) + ((i2 >> 3) << 4)];
        }
        #pragma unroll
        for (int t = 0; t < 4; t++)
            #pragma unroll
            for (int u = 0; u < 2; u++)
                acc[t][u] = __builtin_amdgcn_wmma_f32_16x16x32_bf16(
                    false, af[t], false, bfr[u], (short)0, acc[t][u], false, false);

        __syncthreads();                        // all waves done with buffer `cur`
        __builtin_amdgcn_s_wait_tensorcnt(0);   // issuer waits for prefetch (no-op elsewhere)
        __syncthreads();                        // release: next buffer is ready
    }

    // store: D layout -> m = r + 8*lane[4], n = lane[3:0]
    const int cH = lane >> 4;
    const int cN = lane & 15;
    #pragma unroll
    for (int t = 0; t < 4; t++) {
        const int rowBase = m0 + wr * 64 + t * 16 + 8 * cH;
        #pragma unroll
        for (int u = 0; u < 2; u++) {
            const int col = n0 + wc * 32 + u * 16 + cN;
            const float bv = bias[col];
            #pragma unroll
            for (int r = 0; r < 8; r++)
                C[(size_t)(rowBase + r) * N + col] = acc[t][u][r] + bv;
        }
    }
}

// ---------------------------------------------------------------------------
// 3) Depthwise conv (k=3, zero pad) on u-slot of proj + transpose to (B*D, S)
// ---------------------------------------------------------------------------
__global__ void conv_transpose_kernel(const float* __restrict__ proj,
                                      const float* __restrict__ conv_w,
                                      const float* __restrict__ conv_b,
                                      float* __restrict__ ut) {
    __shared__ float tile[32][33];
    const int b  = blockIdx.z;
    const int d0 = blockIdx.y * 32;
    const int s0 = blockIdx.x * 32;
    const int tx = threadIdx.x, ty = threadIdx.y;
    const int d = d0 + tx;
    const float w0 = conv_w[d * 3 + 0], w1 = conv_w[d * 3 + 1],
                w2 = conv_w[d * 3 + 2], cb = conv_b[d];
    for (int i = ty; i < 32; i += 8) {
        const int s = s0 + i;
        const size_t base = ((size_t)(b * SEQ + s)) * (3 * DIMV) + DIMV + d;
        const float um = (s > 0)       ? proj[base - 3 * DIMV] : 0.f;
        const float uc = proj[base];
        const float up = (s < SEQ - 1) ? proj[base + 3 * DIMV] : 0.f;
        tile[i][tx] = um * w0 + uc * w1 + up * w2 + cb;
    }
    __syncthreads();
    for (int i = ty; i < 32; i += 8)
        ut[((size_t)(b * DIMV + d0 + i)) * SEQ + s0 + tx] = tile[tx][i];
}

// ---------------------------------------------------------------------------
// 4) Stockham radix-2 complex FFT (N=8192) in LDS ping-pong buffers.
// ---------------------------------------------------------------------------
__device__ void fft8192(float2* __restrict__ x, float2* __restrict__ y,
                        int tid, float sgn) {
    float2* src = x;
    float2* dst = y;
    int n = SEQ, ls = 0;
    while (n > 1) {
        const int m = n >> 1;
        const int s = 1 << ls;
        const float theta0 = sgn * 6.28318530717958648f / (float)n;
        for (int t = tid; t < SEQ / 2; t += 256) {
            const int q = t & (s - 1);
            const int p = t >> ls;
            float sw, cw;
            __sincosf(theta0 * (float)p, &sw, &cw);
            const float2 a = src[q + s * p];
            const float2 b = src[q + s * (p + m)];
            float2 e; e.x = a.x + b.x; e.y = a.y + b.y;
            const float dx = a.x - b.x, dy = a.y - b.y;
            float2 o; o.x = dx * cw - dy * sw; o.y = dx * sw + dy * cw;
            dst[q + s * (2 * p)]     = e;
            dst[q + s * (2 * p + 1)] = o;
        }
        __syncthreads();
        float2* tp = src; src = dst; dst = tp;
        n = m; ls++;
    }
}

__global__ __launch_bounds__(256)
void fft_filter_kernel(const float* __restrict__ filt_t, float2* __restrict__ F) {
    extern __shared__ float2 sm[];
    float2* A  = sm;
    float2* Bb = sm + SEQ;
    const int d = blockIdx.x;
    const float* src = filt_t + (size_t)d * SEQ;
    for (int i = threadIdx.x; i < SEQ; i += 256) { A[i].x = src[i]; A[i].y = 0.f; }
    __syncthreads();
    fft8192(A, Bb, threadIdx.x, -1.0f);
    float2* out = F + (size_t)d * SEQ;
    for (int i = threadIdx.x; i < SEQ; i += 256) out[i] = Bb[i];
}

__global__ __launch_bounds__(256)
void fft_conv_kernel(float* __restrict__ ut, const float2* __restrict__ F) {
    extern __shared__ float2 sm[];
    float2* A  = sm;
    float2* Bb = sm + SEQ;
    const int row = blockIdx.x;          // b*D + d
    const int d   = row & (DIMV - 1);
    float* sig = ut + (size_t)row * SEQ;
    for (int i = threadIdx.x; i < SEQ; i += 256) { A[i].x = sig[i]; A[i].y = 0.f; }
    __syncthreads();
    fft8192(A, Bb, threadIdx.x, -1.0f);
    const float2* Fd = F + (size_t)d * SEQ;
    for (int i = threadIdx.x; i < SEQ; i += 256) {
        const float2 u = Bb[i], f = Fd[i];
        float2 r; r.x = u.x * f.x - u.y * f.y; r.y = u.x * f.y + u.y * f.x;
        Bb[i] = r;
    }
    __syncthreads();
    fft8192(Bb, A, threadIdx.x, 1.0f);
    const float inv = 1.0f / (float)SEQ;
    for (int i = threadIdx.x; i < SEQ; i += 256) sig[i] = A[i].x * inv;
}

// ---------------------------------------------------------------------------
// 5) gating v * filtered * sigmoid(z), transpose (B*D,S)->(B*S,D), emit bf16
// ---------------------------------------------------------------------------
__global__ void gate_transpose_kernel(const float* __restrict__ proj,
                                      const float* __restrict__ yt,
                                      __bf16* __restrict__ gated) {
    __shared__ float tile[32][33];
    const int b  = blockIdx.z;
    const int d0 = blockIdx.y * 32;
    const int s0 = blockIdx.x * 32;
    const int tx = threadIdx.x, ty = threadIdx.y;
    for (int i = ty; i < 32; i += 8)
        tile[i][tx] = yt[((size_t)(b * DIMV + d0 + i)) * SEQ + s0 + tx];
    __syncthreads();
    for (int i = ty; i < 32; i += 8) {
        const int s = s0 + i, d = d0 + tx;
        const size_t pb = ((size_t)(b * SEQ + s)) * (3 * DIMV);
        const float v   = proj[pb + d];
        const float z   = proj[pb + 2 * DIMV + d];
        const float sig = 1.f / (1.f + __expf(-z));
        gated[((size_t)(b * SEQ + s)) * DIMV + d] = to_bf16(v * tile[tx][i] * sig);
    }
}

// ---------------------------------------------------------------------------
// launch
// ---------------------------------------------------------------------------
extern "C" void kernel_launch(void* const* d_in, const int* in_sizes, int n_in,
                              void* d_out, int out_size, void* d_ws, size_t ws_size,
                              hipStream_t stream) {
    const float* x      = (const float*)d_in[0];
    const float* w1     = (const float*)d_in[1];
    const float* b1     = (const float*)d_in[2];
    const float* w2     = (const float*)d_in[3];
    const float* b2     = (const float*)d_in[4];
    const float* w3     = (const float*)d_in[5];
    const float* b3     = (const float*)d_in[6];
    const float* conv_w = (const float*)d_in[7];
    const float* conv_b = (const float*)d_in[8];
    const float* w_in   = (const float*)d_in[9];
    const float* b_in   = (const float*)d_in[10];
    const float* w_out  = (const float*)d_in[11];
    const float* b_out  = (const float*)d_in[12];
    float* out = (float*)d_out;

    const int B = 4, M = B * SEQ;               // 32768 rows
    char* ws = (char*)d_ws;
    float*  proj     = (float*)(ws);                                 // 201.3 MB
    float*  filt_t   = (float*)(ws + (size_t)201326592);             // 16.8 MB
    float2* F        = (float2*)(ws + (size_t)218103808);            // 33.6 MB
    float*  ut       = (float*)(ws + (size_t)251658240);             // 67.1 MB
    __bf16* gated    = (__bf16*)(ws + (size_t)318767104);            // 33.6 MB
    __bf16* x_bf     = (__bf16*)(ws + (size_t)352321536);            // 33.6 MB
    __bf16* w_in_t   = (__bf16*)(ws + (size_t)385875968);            // 1.5 MB (1536x512)
    __bf16* w_out_t  = (__bf16*)(ws + (size_t)387448832);            // 0.5 MB (512x512)

    // 0) precision prep: bf16 activations, transposed bf16 weights (N x K)
    cvt_bf16_kernel<<<(M * DIMV / 4 + 255) / 256, 256, 0, stream>>>(x, x_bf, M * DIMV / 4);
    {
        dim3 g1t((3 * DIMV) / 32, DIMV / 32);
        transpose_bf16_kernel<<<g1t, dim3(32, 8), 0, stream>>>(w_in, w_in_t, DIMV, 3 * DIMV);
        dim3 g2t(DIMV / 32, DIMV / 32);
        transpose_bf16_kernel<<<g2t, dim3(32, 8), 0, stream>>>(w_out, w_out_t, DIMV, DIMV);
    }

    // 1) implicit filter MLP -> filt_t (D,S)
    filter_mlp_kernel<<<SEQ / 32, 256, 0, stream>>>(w1, b1, w2, b2, w3, b3, filt_t);

    // 2) filter spectrum
    fft_filter_kernel<<<DIMV, 256, 2 * SEQ * sizeof(float2), stream>>>(filt_t, F);

    // 3) in_proj GEMM: proj = x @ w_in + b_in   (32768 x 1536, K=512)
    {
        dim3 grid((3 * DIMV) / 128, M / 128);
        gemm_bf16_wmma_tdm<<<grid, 256, 0, stream>>>(x_bf, w_in_t, b_in, proj,
                                                     M, 3 * DIMV, DIMV);
    }

    // 4) depthwise conv on u + transpose -> ut (B*D, S)
    {
        dim3 grid(SEQ / 32, DIMV / 32, B);
        conv_transpose_kernel<<<grid, dim3(32, 8), 0, stream>>>(proj, conv_w, conv_b, ut);
    }

    // 5) FFT circular convolution, in place on ut
    fft_conv_kernel<<<B * DIMV, 256, 2 * SEQ * sizeof(float2), stream>>>(ut, F);

    // 6) gating + transpose back -> gated bf16 (B*S, D)
    {
        dim3 grid(SEQ / 32, DIMV / 32, B);
        gate_transpose_kernel<<<grid, dim3(32, 8), 0, stream>>>(proj, ut, gated);
    }

    // 7) out_proj GEMM: out = gated @ w_out + b_out  (32768 x 512, K=512)
    {
        dim3 grid(DIMV / 128, M / 128);
        gemm_bf16_wmma_tdm<<<grid, 256, 0, stream>>>(gated, w_out_t, b_out, out,
                                                     M, DIMV, DIMV);
    }
}
